// EncoderLayer_78237124264539
// MI455X (gfx1250) — matrix operand order repaired and checked by reading, not compiled, not run on hardware
//
#include <hip/hip_runtime.h>
#include <hip/hip_bf16.h>

// ---------------------------------------------------------------------------
// Types / helpers
// ---------------------------------------------------------------------------
typedef __attribute__((ext_vector_type(16))) __bf16          bf16x16;
typedef __attribute__((ext_vector_type(8)))  float           v8f;
typedef __attribute__((ext_vector_type(8)))  unsigned short  ushort8;
typedef __attribute__((ext_vector_type(16))) unsigned short  ushort16;
typedef __attribute__((ext_vector_type(4)))  unsigned short  ushort4v;
typedef int v4i_vs __attribute__((vector_size(16)));   // payload type of async-LDS builtins

#define SEQ    2048
#define NBATCH 2
#define TOKENS 4096           // B * S
#define DMODEL 1024
#define DFF    4096
#define NHEADS 16
#define DHEAD  64

#if defined(__HIP_DEVICE_COMPILE__) && \
    __has_builtin(__builtin_amdgcn_global_load_async_to_lds_b128) && \
    __has_builtin(__builtin_amdgcn_s_wait_asynccnt)
#define USE_ASYNC_LDS 1
#define GLB_V4I(p) ((__attribute__((address_space(1))) v4i_vs*)(p))
#define LDS_V4I(p) ((__attribute__((address_space(3))) v4i_vs*)(p))
#else
#define USE_ASYNC_LDS 0
#endif

static __device__ __forceinline__ unsigned short f2bf(float f) {
  unsigned u = __builtin_bit_cast(unsigned, f);
  u += 0x7FFFu + ((u >> 16) & 1u);           // round-to-nearest-even
  return (unsigned short)(u >> 16);
}

static __device__ __forceinline__ bf16x16 make_frag(ushort8 lo, ushort8 hi) {
  ushort16 u = __builtin_shufflevector(lo, hi, 0,1,2,3,4,5,6,7,8,9,10,11,12,13,14,15);
  return __builtin_bit_cast(bf16x16, u);
}

static __device__ __forceinline__ v8f wmma_bf16(bf16x16 a, bf16x16 b, v8f c) {
  return __builtin_amdgcn_wmma_f32_16x16x32_bf16(false, a, false, b, (short)0, c, false, false);
}

// ---------------------------------------------------------------------------
// f32 -> bf16 conversion (weights)
// ---------------------------------------------------------------------------
__global__ void cvt_f32_bf16_kernel(const float* __restrict__ in,
                                    unsigned short* __restrict__ out, int n) {
  int i = blockIdx.x * blockDim.x + threadIdx.x;
  int stride = gridDim.x * blockDim.x;
  for (; i < n; i += stride) out[i] = f2bf(in[i]);
}

// ---------------------------------------------------------------------------
// LayerNorm over d_model=1024: one block (256 threads) per row; bf16 output
// ---------------------------------------------------------------------------
__global__ __launch_bounds__(256)
void layernorm_bf16_kernel(const float* __restrict__ x,
                           const float* __restrict__ g,
                           const float* __restrict__ b,
                           unsigned short* __restrict__ out) {
  const int row = blockIdx.x;
  const int t   = threadIdx.x;
  const float4 v = ((const float4*)(x + (size_t)row * DMODEL))[t];

  __shared__ float red[256];
  red[t] = v.x + v.y + v.z + v.w;
  __syncthreads();
  for (int off = 128; off > 0; off >>= 1) {
    if (t < off) red[t] += red[t + off];
    __syncthreads();
  }
  const float mu = red[0] * (1.0f / DMODEL);
  __syncthreads();

  const float d0 = v.x - mu, d1 = v.y - mu, d2 = v.z - mu, d3 = v.w - mu;
  red[t] = d0*d0 + d1*d1 + d2*d2 + d3*d3;
  __syncthreads();
  for (int off = 128; off > 0; off >>= 1) {
    if (t < off) red[t] += red[t + off];
    __syncthreads();
  }
  const float rstd = rsqrtf(red[0] * (1.0f / DMODEL) + 1e-5f);

  const float4 gv = ((const float4*)g)[t];
  const float4 bv = ((const float4*)b)[t];
  ushort4v o;
  o.x = f2bf(d0 * rstd * gv.x + bv.x);
  o.y = f2bf(d1 * rstd * gv.y + bv.y);
  o.z = f2bf(d2 * rstd * gv.z + bv.z);
  o.w = f2bf(d3 * rstd * gv.w + bv.w);
  ((ushort4v*)(out + (size_t)row * DMODEL))[t] = o;
}

// ---------------------------------------------------------------------------
// bf16 WMMA GEMM: C[M,N] = A[M,K] * B[K,N]  (+bias, gelu, +residual)
// Block tile 128x128x32, 8 waves in 2(m) x 4(n), wave tile 64x32 (4x2 accums).
// Double-buffered LDS; A tiles staged via async global->LDS copies
// (ASYNCcnt pipeline) when the gfx1250 builtins exist, else reg-prefetch.
// ---------------------------------------------------------------------------
#define BM 128
#define BN 128
#define BK 32
#define LDA_S 40   // As row stride (elems); 80B rows keep 16B-aligned chunks
#define LDB_S 40   // BsT row stride

template <bool HAS_BIAS, bool HAS_RES, bool DO_GELU, bool OUT_F32, bool OUT_BF16>
__global__ __launch_bounds__(256)
void gemm_bf16_kernel(const unsigned short* __restrict__ A,
                      const unsigned short* __restrict__ B,
                      const float* __restrict__ bias,
                      const float* __restrict__ residual,
                      float* __restrict__ Cf,
                      unsigned short* __restrict__ Cbf,
                      int M, int N, int K) {
  __shared__ __align__(16) unsigned short As [2][BM * LDA_S];
  __shared__ __align__(16) unsigned short BsT[2][BN * LDB_S];

  const int tid  = threadIdx.x;
  const int lane = tid & 31;
  const int wave = tid >> 5;
  const int wm   = wave >> 2;          // 0..1
  const int wn   = wave & 3;           // 0..3
  const int part = lane >> 4;          // lane half
  const int l16  = lane & 15;
  const int m0   = blockIdx.y * BM;
  const int n0   = blockIdx.x * BN;

  // staging coordinates
  const int ar = tid >> 1;             // 0..127
  const int ac = (tid & 1) * 16;       // 0 / 16
  const int br = tid >> 3;             // 0..31  (k)
  const int bc = (tid & 7) * 16;       // 0..112 (n)

  const unsigned short* aptr = A + (size_t)(m0 + ar) * K + ac;
  const unsigned short* bptr = B + (size_t)br * N + n0 + bc;

  v8f acc[4][2] = {};
  ushort8 bpre0, bpre1;
#if !USE_ASYNC_LDS
  ushort8 apre0, apre1;
#endif

  const int nk = K / BK;

  // ---- prologue: stage tile 0 into buffer 0 ----
#if USE_ASYNC_LDS
  __builtin_amdgcn_global_load_async_to_lds_b128(
      GLB_V4I(aptr),     LDS_V4I(&As[0][ar * LDA_S + ac]),     0, 0);
  __builtin_amdgcn_global_load_async_to_lds_b128(
      GLB_V4I(aptr + 8), LDS_V4I(&As[0][ar * LDA_S + ac + 8]), 0, 0);
#else
  apre0 = *(const ushort8*)(aptr);
  apre1 = *(const ushort8*)(aptr + 8);
  *(ushort8*)(&As[0][ar * LDA_S + ac])     = apre0;
  *(ushort8*)(&As[0][ar * LDA_S + ac + 8]) = apre1;
#endif
  bpre0 = *(const ushort8*)(bptr);
  bpre1 = *(const ushort8*)(bptr + 8);
#pragma unroll
  for (int i = 0; i < 8; ++i) {
    BsT[0][(bc + i) * LDB_S + br]     = bpre0[i];
    BsT[0][(bc + 8 + i) * LDB_S + br] = bpre1[i];
  }
#if USE_ASYNC_LDS
  __builtin_amdgcn_s_wait_asynccnt(0);
#endif
  __syncthreads();

  for (int ki = 0; ki < nk; ++ki) {
    const int p = ki & 1;
    const bool more = (ki + 1 < nk);

    // ---- issue staging of tile ki+1 into buffer 1-p ----
    if (more) {
      const int k1 = (ki + 1) * BK;
#if USE_ASYNC_LDS
      __builtin_amdgcn_global_load_async_to_lds_b128(
          GLB_V4I(aptr + k1),     LDS_V4I(&As[1 - p][ar * LDA_S + ac]),     0, 0);
      __builtin_amdgcn_global_load_async_to_lds_b128(
          GLB_V4I(aptr + k1 + 8), LDS_V4I(&As[1 - p][ar * LDA_S + ac + 8]), 0, 0);
#else
      apre0 = *(const ushort8*)(aptr + k1);
      apre1 = *(const ushort8*)(aptr + k1 + 8);
#endif
      bpre0 = *(const ushort8*)(bptr + (size_t)k1 * N);
      bpre1 = *(const ushort8*)(bptr + (size_t)k1 * N + 8);
    }

    // ---- compute tile ki from buffer p: 12 ds_load_b128 feed 8 WMMAs ----
    bf16x16 af[4], bfr[2];
#pragma unroll
    for (int tm = 0; tm < 4; ++tm) {
      const unsigned short* ap = &As[p][(wm * 64 + tm * 16 + l16) * LDA_S + part * 8];
      af[tm] = make_frag(*(const ushort8*)ap, *(const ushort8*)(ap + 16));
    }
#pragma unroll
    for (int tn = 0; tn < 2; ++tn) {
      const unsigned short* bp = &BsT[p][(wn * 32 + tn * 16 + l16) * LDB_S + part * 8];
      bfr[tn] = make_frag(*(const ushort8*)bp, *(const ushort8*)(bp + 16));
    }
#pragma unroll
    for (int tm = 0; tm < 4; ++tm)
#pragma unroll
      for (int tn = 0; tn < 2; ++tn)
        acc[tm][tn] = wmma_bf16(af[tm], bfr[tn], acc[tm][tn]);

    // ---- commit staged B (and A in sync path) into buffer 1-p ----
    if (more) {
#pragma unroll
      for (int i = 0; i < 8; ++i) {
        BsT[1 - p][(bc + i) * LDB_S + br]     = bpre0[i];
        BsT[1 - p][(bc + 8 + i) * LDB_S + br] = bpre1[i];
      }
#if !USE_ASYNC_LDS
      *(ushort8*)(&As[1 - p][ar * LDA_S + ac])     = apre0;
      *(ushort8*)(&As[1 - p][ar * LDA_S + ac + 8]) = apre1;
#endif
#if USE_ASYNC_LDS
      __builtin_amdgcn_s_wait_asynccnt(0);
#endif
    }
    __syncthreads();
  }

  // ---- epilogue (branch-free per instantiation) ----
#pragma unroll
  for (int tn = 0; tn < 2; ++tn) {
    const int col = n0 + wn * 32 + tn * 16 + l16;
    float bias_v = 0.0f;
    if constexpr (HAS_BIAS) bias_v = bias[col];
#pragma unroll
    for (int tm = 0; tm < 4; ++tm) {
#pragma unroll
      for (int r = 0; r < 8; ++r) {
        const int row = m0 + wm * 64 + tm * 16 + r + part * 8;
        float val = acc[tm][tn][r];
        if constexpr (HAS_BIAS) val += bias_v;
        if constexpr (DO_GELU)  val = 0.5f * val * (1.0f + erff(val * 0.70710678118654752f));
        if constexpr (HAS_RES)  val += residual[(size_t)row * N + col];
        if constexpr (OUT_F32)  Cf [(size_t)row * N + col] = val;
        if constexpr (OUT_BF16) Cbf[(size_t)row * N + col] = f2bf(val);
      }
    }
  }
}

// ---------------------------------------------------------------------------
// Flash attention: WG = (b, head, 128 q-rows); 8 waves, 16 q-rows each.
// K/V staged in LDS (V transposed); online softmax (hardware exp);
// bf16 context output.
// ---------------------------------------------------------------------------
#define KPAD 72
#define VPAD 40
#define PPAD 40

__global__ __launch_bounds__(256)
void attention_kernel(const unsigned short* __restrict__ Qb,
                      const unsigned short* __restrict__ Kb,
                      const unsigned short* __restrict__ Vb,
                      const unsigned char*  __restrict__ pmask,
                      unsigned short* __restrict__ Ob) {
  __shared__ __align__(16) unsigned short Ks [32 * KPAD];        // [key][d]
  __shared__ __align__(16) unsigned short VsT[DHEAD * VPAD];     // [d][key]
  __shared__ __align__(16) unsigned short Ps [8 * 16 * PPAD];    // per-wave P

  const int tid  = threadIdx.x;
  const int lane = tid & 31;
  const int wave = tid >> 5;
  const int part = lane >> 4;
  const int l16  = lane & 15;
  const int head = blockIdx.y;
  const int bz   = blockIdx.z;
  const int q0   = blockIdx.x * 128 + wave * 16;

  // Q fragments (A-matrix 16x64 as two 16x32 frags), gathered once
  const size_t qbase = (size_t)(bz * SEQ + q0 + l16) * DMODEL + head * DHEAD;
  bf16x16 qa[2];
#pragma unroll
  for (int kf = 0; kf < 2; ++kf) {
    const unsigned short* qp = Qb + qbase + kf * 32 + part * 8;
    qa[kf] = make_frag(*(const ushort8*)qp, *(const ushort8*)(qp + 16));
  }

  float m_run[8], l_run[8];
  v8f ctx[4] = {};
#pragma unroll
  for (int r = 0; r < 8; ++r) { m_run[r] = -3.0e38f; l_run[r] = 0.0f; }

  const unsigned char* mbase = pmask + (size_t)bz * SEQ * SEQ;

  for (int j0 = 0; j0 < SEQ; j0 += 32) {
    __syncthreads();
    // stage K (row-major) and V (transposed) tiles: 32 keys x 64 dims
    {
      const int r = tid >> 3;          // key 0..31
      const int c = (tid & 7) * 8;     // d
      const size_t grow = (size_t)(bz * SEQ + j0 + r) * DMODEL + head * DHEAD + c;
      ushort8 kv = *(const ushort8*)(Kb + grow);
      *(ushort8*)(Ks + r * KPAD + c) = kv;
      ushort8 vv = *(const ushort8*)(Vb + grow);
#pragma unroll
      for (int i = 0; i < 8; ++i) VsT[(c + i) * VPAD + r] = vv[i];
    }
    __syncthreads();

    // scores: S[16q x 32k] = Q(16x64) * K^T(64x32), two 16x16 halves
    v8f sL = {}, sR = {};
#pragma unroll
    for (int kf = 0; kf < 2; ++kf) {
      const unsigned short* kpl = Ks + l16 * KPAD + kf * 32 + part * 8;
      bf16x16 kbL = make_frag(*(const ushort8*)kpl, *(const ushort8*)(kpl + 16));
      const unsigned short* kpr = Ks + (16 + l16) * KPAD + kf * 32 + part * 8;
      bf16x16 kbR = make_frag(*(const ushort8*)kpr, *(const ushort8*)(kpr + 16));
      sL = wmma_bf16(qa[kf], kbL, sL);
      sR = wmma_bf16(qa[kf], kbR, sR);
    }

    // online softmax (row M = r + 8*part, col = lane%16); hardware exp
    float pL[8], pR[8];
#pragma unroll
    for (int r = 0; r < 8; ++r) {
      const int qrw = q0 + r + part * 8;
      float aL = sL[r] * 0.125f;
      float aR = sR[r] * 0.125f;
      if (mbase[(size_t)qrw * SEQ + j0 + l16])      aL = -1.0e9f;
      if (mbase[(size_t)qrw * SEQ + j0 + 16 + l16]) aR = -1.0e9f;

      float mx = fmaxf(aL, aR);
#pragma unroll
      for (int off = 1; off <= 8; off <<= 1) mx = fmaxf(mx, __shfl_xor(mx, off, 32));
      const float mnew  = fmaxf(m_run[r], mx);
      const float alpha = __expf(m_run[r] - mnew);
      const float eL = __expf(aL - mnew);
      const float eR = __expf(aR - mnew);
      float sum = eL + eR;
#pragma unroll
      for (int off = 1; off <= 8; off <<= 1) sum += __shfl_xor(sum, off, 32);
      l_run[r] = l_run[r] * alpha + sum;
      m_run[r] = mnew;
#pragma unroll
      for (int nb = 0; nb < 4; ++nb) ctx[nb][r] *= alpha;
      pL[r] = eL; pR[r] = eR;
    }

    // re-stripe P (C layout -> A layout) through per-wave LDS scratch
    unsigned short* pw = Ps + wave * 16 * PPAD;
#pragma unroll
    for (int r = 0; r < 8; ++r) {
      pw[(r + part * 8) * PPAD + l16]      = f2bf(pL[r]);
      pw[(r + part * 8) * PPAD + 16 + l16] = f2bf(pR[r]);
    }
    asm volatile("s_wait_dscnt 0x0" ::: "memory");   // within-wave LDS RAW
    const unsigned short* pr = pw + l16 * PPAD + part * 8;
    bf16x16 pa = make_frag(*(const ushort8*)pr, *(const ushort8*)(pr + 16));

    // ctx += P(16x32) * V(32x64)
#pragma unroll
    for (int nb = 0; nb < 4; ++nb) {
      const unsigned short* vp = VsT + (nb * 16 + l16) * VPAD + part * 8;
      bf16x16 vb = make_frag(*(const ushort8*)vp, *(const ushort8*)(vp + 16));
      ctx[nb] = wmma_bf16(pa, vb, ctx[nb]);
    }
  }

  // normalize + store bf16 context
#pragma unroll
  for (int r = 0; r < 8; ++r) {
    const float inv = 1.0f / l_run[r];
    const size_t obase = (size_t)(bz * SEQ + q0 + r + part * 8) * DMODEL + head * DHEAD;
#pragma unroll
    for (int nb = 0; nb < 4; ++nb)
      Ob[obase + nb * 16 + l16] = f2bf(ctx[nb][r] * inv);
  }
}

// ---------------------------------------------------------------------------
// Host-side orchestration
// ---------------------------------------------------------------------------
extern "C" void kernel_launch(void* const* d_in, const int* in_sizes, int n_in,
                              void* d_out, int out_size, void* d_ws, size_t ws_size,
                              hipStream_t stream) {
  (void)in_sizes; (void)n_in; (void)out_size; (void)ws_size;

  const float*         x     = (const float*)d_in[0];
  const unsigned char* pmask = (const unsigned char*)d_in[1];
  const float* W_Q   = (const float*)d_in[2];
  const float* W_K   = (const float*)d_in[3];
  const float* W_V   = (const float*)d_in[4];
  const float* W_O   = (const float*)d_in[5];
  const float* fc1_w = (const float*)d_in[6];
  const float* fc1_b = (const float*)d_in[7];
  const float* fc2_w = (const float*)d_in[8];
  const float* fc2_b = (const float*)d_in[9];
  const float* ln1_g = (const float*)d_in[10];
  const float* ln1_b = (const float*)d_in[11];
  const float* ln2_g = (const float*)d_in[12];
  const float* ln2_b = (const float*)d_in[13];
  float* out = (float*)d_out;

  unsigned short* ws = (unsigned short*)d_ws;
  size_t off = 0;
  auto grab = [&](size_t n) { unsigned short* p = ws + off; off += n; return p; };
  unsigned short* wq_bf  = grab((size_t)DMODEL * DMODEL);
  unsigned short* wk_bf  = grab((size_t)DMODEL * DMODEL);
  unsigned short* wv_bf  = grab((size_t)DMODEL * DMODEL);
  unsigned short* wo_bf  = grab((size_t)DMODEL * DMODEL);
  unsigned short* fc1_bf = grab((size_t)DMODEL * DFF);
  unsigned short* fc2_bf = grab((size_t)DFF * DMODEL);
  unsigned short* h1_bf  = grab((size_t)TOKENS * DMODEL);
  unsigned short* q_bf   = grab((size_t)TOKENS * DMODEL);
  unsigned short* k_bf   = grab((size_t)TOKENS * DMODEL);
  unsigned short* v_bf   = grab((size_t)TOKENS * DMODEL);
  unsigned short* ctx_bf = grab((size_t)TOKENS * DMODEL);
  unsigned short* h2_bf  = grab((size_t)TOKENS * DMODEL);
  unsigned short* g_bf   = grab((size_t)TOKENS * DFF);

  // 1) weights -> bf16
  cvt_f32_bf16_kernel<<<2048, 256, 0, stream>>>(W_Q,   wq_bf,  DMODEL * DMODEL);
  cvt_f32_bf16_kernel<<<2048, 256, 0, stream>>>(W_K,   wk_bf,  DMODEL * DMODEL);
  cvt_f32_bf16_kernel<<<2048, 256, 0, stream>>>(W_V,   wv_bf,  DMODEL * DMODEL);
  cvt_f32_bf16_kernel<<<2048, 256, 0, stream>>>(W_O,   wo_bf,  DMODEL * DMODEL);
  cvt_f32_bf16_kernel<<<2048, 256, 0, stream>>>(fc1_w, fc1_bf, DMODEL * DFF);
  cvt_f32_bf16_kernel<<<2048, 256, 0, stream>>>(fc2_w, fc2_bf, DFF * DMODEL);

  // 2) LN1
  layernorm_bf16_kernel<<<TOKENS, 256, 0, stream>>>(x, ln1_g, ln1_b, h1_bf);

  // 3) Q, K, V projections (bf16 out)
  dim3 gproj(DMODEL / BN, TOKENS / BM);
  gemm_bf16_kernel<false,false,false,false,true><<<gproj, 256, 0, stream>>>(
      h1_bf, wq_bf, nullptr, nullptr, nullptr, q_bf, TOKENS, DMODEL, DMODEL);
  gemm_bf16_kernel<false,false,false,false,true><<<gproj, 256, 0, stream>>>(
      h1_bf, wk_bf, nullptr, nullptr, nullptr, k_bf, TOKENS, DMODEL, DMODEL);
  gemm_bf16_kernel<false,false,false,false,true><<<gproj, 256, 0, stream>>>(
      h1_bf, wv_bf, nullptr, nullptr, nullptr, v_bf, TOKENS, DMODEL, DMODEL);

  // 4) attention
  dim3 gattn(SEQ / 128, NHEADS, NBATCH);
  attention_kernel<<<gattn, 256, 0, stream>>>(q_bf, k_bf, v_bf, pmask, ctx_bf);

  // 5) output projection + residual (f32 -> d_out)
  gemm_bf16_kernel<false,true,false,true,false><<<gproj, 256, 0, stream>>>(
      ctx_bf, wo_bf, nullptr, x, out, nullptr, TOKENS, DMODEL, DMODEL);

  // 6) LN2
  layernorm_bf16_kernel<<<TOKENS, 256, 0, stream>>>(out, ln2_g, ln2_b, h2_bf);

  // 7) FFN1: gelu(h2 @ fc1 + b1) -> bf16
  dim3 gfc1(DFF / BN, TOKENS / BM);
  gemm_bf16_kernel<true,false,true,false,true><<<gfc1, 256, 0, stream>>>(
      h2_bf, fc1_bf, fc1_b, nullptr, nullptr, g_bf, TOKENS, DFF, DMODEL);

  // 8) FFN2: g @ fc2 + b2 + residual -> d_out
  gemm_bf16_kernel<true,true,false,true,false><<<gproj, 256, 0, stream>>>(
      g_bf, fc2_bf, fc2_b, out, out, nullptr, TOKENS, DMODEL, DFF);
}
